// ConvAConnect_25168508355078
// MI455X (gfx1250) — compile-verified, matrix operand order checked by reading, not run on hardware
//
#include <hip/hip_runtime.h>

typedef __attribute__((ext_vector_type(16))) __bf16 v16bf;
typedef __attribute__((ext_vector_type(4)))  __bf16 v4bf;
typedef __attribute__((ext_vector_type(8)))  float  v8f;

#define B_   32
#define H_   64
#define W_   64
#define CIN  128
#define FF   256
#define HW   (H_*W_)
#define TAPS 9
#define HP   66            // padded spatial extent (halo of 1)

#define TM 128
#define TN 128
#define TK 32
#define NCHUNK (TAPS*(CIN/TK))   // 36 K-chunks of 32

union Frag { v16bf v; uint4 q[2]; };

// ---------------------------------------------------------------------------
// Pass 1a: memW_bf16[b,tap,f,c] = bf16(W[tap,c,f] * Werr[b,tap,c,f])
// (transposed so the GEMM B-stage is k-contiguous; LDS 32x32 tile transpose)
// ---------------------------------------------------------------------------
__global__ __launch_bounds__(256)
void build_memw(const float* __restrict__ Wk, const float* __restrict__ Werr,
                __bf16* __restrict__ Wmem)
{
  __shared__ __bf16 tile[32][34];            // +2 pad: avoid bank conflicts
  const int bt  = blockIdx.x;                // b*9 + tap
  const int ft  = blockIdx.y;                // filter tile (8)
  const int ct  = blockIdx.z;                // cin tile (4)
  const int tap = bt % TAPS;
  const int t   = threadIdx.x;

  // read: coalesced along f
  const int cl = t >> 3;                     // 0..31 cin-in-tile
  const int fl = (t & 7) * 4;                // f-in-tile (float4)
  const int c  = ct*32 + cl;
  const int f0 = ft*32 + fl;
  const float4 w4  = *(const float4*)(Wk   + ((size_t)tap*CIN + c)*FF + f0);
  const float4 we4 = *(const float4*)(Werr + ((size_t)bt *CIN + c)*FF + f0);
  tile[fl+0][cl] = (__bf16)(w4.x*we4.x);
  tile[fl+1][cl] = (__bf16)(w4.y*we4.y);
  tile[fl+2][cl] = (__bf16)(w4.z*we4.z);
  tile[fl+3][cl] = (__bf16)(w4.w*we4.w);
  __syncthreads();

  // write: coalesced along c
  const int fr = t >> 3;
  const int cw = (t & 7) * 4;
  v4bf o;
  o[0]=tile[fr][cw]; o[1]=tile[fr][cw+1]; o[2]=tile[fr][cw+2]; o[3]=tile[fr][cw+3];
  *(v4bf*)(Wmem + ((size_t)bt*FF + ft*32 + fr)*CIN + ct*32 + cw) = o;
}

// ---------------------------------------------------------------------------
// Pass 1b: Xpad_bf16[b, 0..65, 0..65, c] = bf16(X) with zero halo
// ---------------------------------------------------------------------------
__global__ __launch_bounds__(128)
void build_xpad(const float* __restrict__ X, __bf16* __restrict__ Xp)
{
  const int b   = blockIdx.y;
  const int row = blockIdx.x * 4 + threadIdx.y;   // 0..4355 = yp*66+xp
  const int yp  = row / HP;
  const int xp  = row - yp*HP;
  const int ys  = yp - 1, xs = xp - 1;
  const int c   = threadIdx.x * 4;

  float4 v = make_float4(0.f, 0.f, 0.f, 0.f);
  if (((unsigned)ys < (unsigned)H_) && ((unsigned)xs < (unsigned)W_))
    v = *(const float4*)(X + (((size_t)b*H_ + ys)*W_ + xs)*CIN + c);
  v4bf o; o[0]=(__bf16)v.x; o[1]=(__bf16)v.y; o[2]=(__bf16)v.z; o[3]=(__bf16)v.w;
  *(v4bf*)(Xp + (((size_t)b*HP + yp)*HP + xp)*CIN + c) = o;
}

// ---------------------------------------------------------------------------
// Pass 2: implicit GEMM per sample, bf16 WMMA, async-to-LDS double buffered
// ---------------------------------------------------------------------------
__global__ __launch_bounds__(256)
void aconnect_gemm_wmma(const __bf16* __restrict__ Xp,
                        const __bf16* __restrict__ Wmem,
                        const float*  __restrict__ bias,
                        const float*  __restrict__ Berr,
                        float*        __restrict__ Out)
{
  __shared__ __align__(16) __bf16 Abuf[2][TM*TK];   // [pixels][k]
  __shared__ __align__(16) __bf16 Bbuf[2][TN*TK];   // [filters][k]

  const int t    = threadIdx.x;
  const int lane = t & 31;
  const int wave = t >> 5;
  const int b    = blockIdx.y;
  const int mt   = blockIdx.x >> 1;
  const int nt   = blockIdx.x & 1;
  const int m0   = mt * TM;
  const int n0   = nt * TN;

  const int waveM = wave & 3;
  const int waveN = wave >> 2;
  const int mBase = waveM * 32;
  const int nBase = waveN * 64;
  const int hi    = (lane >> 4) & 1;
  const int lrow  = lane & 15;

  // perturbed bias, loaded early
  float mb[4];
  #pragma unroll
  for (int j = 0; j < 4; ++j) {
    int f = n0 + nBase + j*16 + lrow;
    mb[j] = bias[f] * Berr[b*FF + f];
  }

  v8f acc[2][4];
  #pragma unroll
  for (int i = 0; i < 2; ++i)
    #pragma unroll
    for (int j = 0; j < 4; ++j)
      acc[i][j] = (v8f)0.0f;

  // ---- loop-invariant per-lane staging offsets ----
  // A: 2 threads per pixel-row, 16 bf16 (32B) each
  const int ar = t >> 1;
  const int ah = t & 1;
  const int p  = m0 + ar;
  const int y  = p >> 6;
  const int x  = p & 63;
  const unsigned voffA = (unsigned)((((unsigned)b*HP*HP + (unsigned)y*HP + (unsigned)x)*CIN
                                     + (unsigned)ah*16) * 2u);   // bytes
  // B: 1 thread per filter half, 16 bf16 (32B) each
  const int bf_ = t & 127;
  const int bh  = t >> 7;
  const unsigned voffB = (unsigned)(((unsigned)bf_*CIN + (unsigned)bh*16) * 2u); // bytes

  const unsigned ldsA[2] = { (unsigned)(size_t)(void*)&Abuf[0][ar*TK + ah*16],
                             (unsigned)(size_t)(void*)&Abuf[1][ar*TK + ah*16] };
  const unsigned ldsB[2] = { (unsigned)(size_t)(void*)&Bbuf[0][bf_*TK + bh*16],
                             (unsigned)(size_t)(void*)&Bbuf[1][bf_*TK + bh*16] };

  auto stage = [&](int kk, int sel) {
    const int tap = kk >> 2;
    const int c0  = (kk & 3) * TK;
    const int kh  = tap / 3;
    const int kw  = tap - kh*3;
    // uniform (SGPR) bases for this chunk
    const __bf16* pa = Xp   + ((size_t)(kh*HP + kw))*CIN + c0;
    const __bf16* pb = Wmem + ((size_t)(b*TAPS + tap)*FF + n0)*CIN + c0;
    // async copy: 2x b128 each for A and B halves (offset advances both sides)
    asm volatile(
      "global_load_async_to_lds_b128 %0, %1, %2\n\t"
      "global_load_async_to_lds_b128 %0, %1, %2 offset:16"
      :: "v"(ldsA[sel]), "v"(voffA), "s"(pa) : "memory");
    asm volatile(
      "global_load_async_to_lds_b128 %0, %1, %2\n\t"
      "global_load_async_to_lds_b128 %0, %1, %2 offset:16"
      :: "v"(ldsB[sel]), "v"(voffB), "s"(pb) : "memory");
  };

  auto compute = [&](int sel) {
    const uint4* Aq = (const uint4*)Abuf[sel];
    const uint4* Bq = (const uint4*)Bbuf[sel];
    Frag afr[2], bfr[4];
    #pragma unroll
    for (int i = 0; i < 2; ++i) {
      int row  = mBase + i*16 + lrow;
      int qidx = row*4 + hi;            // 64B rows, K-half select
      afr[i].q[0] = Aq[qidx];           // K = kb..kb+7
      afr[i].q[1] = Aq[qidx + 2];       // K = kb+16..kb+23
    }
    #pragma unroll
    for (int j = 0; j < 4; ++j) {
      int col  = nBase + j*16 + lrow;
      int qidx = col*4 + hi;
      bfr[j].q[0] = Bq[qidx];
      bfr[j].q[1] = Bq[qidx + 2];
    }
    #pragma unroll
    for (int i = 0; i < 2; ++i)
      #pragma unroll
      for (int j = 0; j < 4; ++j)
        acc[i][j] = __builtin_amdgcn_wmma_f32_16x16x32_bf16(
            false, afr[i].v, false, bfr[j].v, (short)0, acc[i][j], false, false);
  };

  stage(0, 0);
  for (int kk = 0; kk < NCHUNK; ++kk) {
    const int sel = kk & 1;
    asm volatile("s_wait_asynccnt 0x0" ::: "memory");  // this wave's copies landed
    __syncthreads();                                   // all waves' copies visible
    if (kk + 1 < NCHUNK) stage(kk + 1, sel ^ 1);       // overlap next copy w/ WMMA
    compute(sel);
  }

  // epilogue: bias + ReLU, NHWC store
  const int rowOff = hi ? 8 : 0;
  #pragma unroll
  for (int i = 0; i < 2; ++i) {
    #pragma unroll
    for (int j = 0; j < 4; ++j) {
      const int f = n0 + nBase + j*16 + lrow;
      #pragma unroll
      for (int r = 0; r < 8; ++r) {
        const int pix  = m0 + mBase + i*16 + r + rowOff;
        const float vo = acc[i][j][r] + mb[j];
        Out[((size_t)b*HW + pix)*FF + f] = vo > 0.0f ? vo : 0.0f;
      }
    }
  }
}

// ---------------------------------------------------------------------------
extern "C" void kernel_launch(void* const* d_in, const int* in_sizes, int n_in,
                              void* d_out, int out_size, void* d_ws, size_t ws_size,
                              hipStream_t stream) {
  const float* X    = (const float*)d_in[0];
  const float* Wk   = (const float*)d_in[1];
  const float* bias = (const float*)d_in[2];
  const float* Werr = (const float*)d_in[3];
  const float* Berr = (const float*)d_in[4];
  float* Out = (float*)d_out;

  // workspace layout (bf16): memW [B,9,F,C] then padded X [B,66,66,C]
  // bytes: 18,874,368 + 35,684,352 ~= 54.6 MB
  __bf16* Wmem = (__bf16*)d_ws;
  __bf16* Xp   = (__bf16*)d_ws + (size_t)B_*TAPS*FF*CIN;

  build_memw<<<dim3(B_*TAPS, FF/32, CIN/32), dim3(256), 0, stream>>>(Wk, Werr, Wmem);
  build_xpad<<<dim3((HP*HP)/4, B_), dim3(32, 4), 0, stream>>>(X, Xp);
  aconnect_gemm_wmma<<<dim3((HW/TM)*(FF/TN), B_), dim3(256), 0, stream>>>(
      Xp, Wmem, bias, Berr, Out);

  (void)in_sizes; (void)n_in; (void)out_size; (void)ws_size;
}
